// PointMAEPatchClassifier_15333033247031
// MI455X (gfx1250) — matrix operand order repaired and checked by reading, not compiled, not run on hardware
//
#include <hip/hip_runtime.h>
#include <hip/hip_bf16.h>
#include <math.h>

// ---------------- model dims ----------------
#define BB   64
#define NN   2048
#define GG   64
#define KK   32
#define DD   384
#define HH   6
#define HD   64
#define DEPTH 12
#define NCLS 40
#define NTOK 65            // G+1
#define ATT_SCALE 0.125f   // HD^-0.5

typedef __attribute__((ext_vector_type(16))) _Float16 v16h;
typedef __attribute__((ext_vector_type(8)))  float    v8f;

// =====================================================================
// Weight pre-pass: W (K x N, f32) -> WT (N x Kpad, f16), zero-padded K.
// =====================================================================
__global__ __launch_bounds__(256)
void cvt_wt_k(const float* __restrict__ W, _Float16* __restrict__ WT,
              int K, int N, int Kpad)
{
    int i = blockIdx.x * 256 + threadIdx.x;
    if (i >= N * Kpad) return;
    int n = i / Kpad, k = i % Kpad;
    WT[i] = (k < K) ? (_Float16)W[(size_t)k * N + n] : (_Float16)0.0f;
}

// =====================================================================
// FAST WMMA GEMM: requires M%64==0, N%128==0, Kd%32==0 (all hot layers).
// Block tile 64(M) x 128(N), 8 waves in 2x4; each wave computes 32x32:
// 2 A frags x 2 B frags -> 4 v_wmma_f32_16x16x32_f16 per K-step.
// Double-buffered LDS staging (global b128 loads overlap WMMA).
// =====================================================================
__global__ __launch_bounds__(256)
void gemm_wmma_fast_k(const float* __restrict__ A, const _Float16* __restrict__ WT,
                      const float* __restrict__ bias,
                      const float* __restrict__ bng, const float* __restrict__ bnb,
                      const float* __restrict__ bnm, const float* __restrict__ bnv,
                      const float* __restrict__ residual,
                      float* __restrict__ C,
                      int M, int N, int Kd, int act)
{
    __shared__ _Float16 As[2][64 * 40];    // 5 KB each
    __shared__ _Float16 Bs[2][128 * 40];   // 10 KB each

    const int tid  = threadIdx.x;
    const int lane = tid & 31;
    const int wid  = tid >> 5;
    const int wm   = wid & 1;              // 2 wave-rows x 32 M
    const int wn   = wid >> 1;             // 4 wave-cols x 32 N
    const int half = lane >> 4;
    const int r16  = lane & 15;
    const int rowBase = blockIdx.y * 64;
    const int colBase = blockIdx.x * 128;

    // staging assignments
    const int ar = tid >> 2;               // 0..63 (A row)
    const int ac = (tid & 3) * 8;          // 0,8,16,24 (A col, 8 floats)
    const int bn = tid >> 1;               // 0..127 (B col n)
    const int bp = (tid & 1) * 16;         // 0,16 (16 halfs)

    auto stageA = [&](int kk, int buf) {
        const float* src = A + (size_t)(rowBase + ar) * Kd + kk + ac;
        float4 f0 = *(const float4*)src;
        float4 f1 = *(const float4*)(src + 4);
        union { _Float16 h[8]; uint4 u; } pk;
        pk.h[0] = (_Float16)f0.x; pk.h[1] = (_Float16)f0.y;
        pk.h[2] = (_Float16)f0.z; pk.h[3] = (_Float16)f0.w;
        pk.h[4] = (_Float16)f1.x; pk.h[5] = (_Float16)f1.y;
        pk.h[6] = (_Float16)f1.z; pk.h[7] = (_Float16)f1.w;
        *(uint4*)&As[buf][ar * 40 + ac] = pk.u;
        if (kk + 64 < Kd) __builtin_prefetch(src + 64, 0, 1);
    };
    auto stageB = [&](int kk, int buf) {
        const _Float16* src = &WT[(size_t)(colBase + bn) * Kd + kk + bp];
        uint4 u0 = *(const uint4*)src;
        uint4 u1 = *(const uint4*)(src + 8);
        *(uint4*)&Bs[buf][bn * 40 + bp]     = u0;
        *(uint4*)&Bs[buf][bn * 40 + bp + 8] = u1;
    };

    stageA(0, 0);
    stageB(0, 0);
    __syncthreads();

    v8f acc[2][2] = {};
    const int nsteps = Kd >> 5;
    for (int s = 0; s < nsteps; ++s) {
        const int cur = s & 1;
        if (s + 1 < nsteps) {               // overlap next-tile loads with WMMA
            stageA((s + 1) * 32, cur ^ 1);
            stageB((s + 1) * 32, cur ^ 1);
        }
        union { v16h h; unsigned int u[8]; } af[2], bf[2];
        #pragma unroll
        for (int i = 0; i < 2; ++i) {
            const int arow = wm * 32 + i * 16 + r16;
            #pragma unroll
            for (int vv = 0; vv < 4; ++vv)
                af[i].u[vv] = *(const unsigned int*)&As[cur][arow * 40 + half * 8 + 2 * vv];
            #pragma unroll
            for (int vv = 4; vv < 8; ++vv)
                af[i].u[vv] = *(const unsigned int*)&As[cur][arow * 40 + 16 + half * 8 + 2 * (vv - 4)];
            const int bcol = wn * 32 + i * 16 + r16;
            #pragma unroll
            for (int vv = 0; vv < 8; ++vv)
                bf[i].u[vv] = *(const unsigned int*)&Bs[cur][bcol * 40 + (vv >> 2) * 16 + half * 8 + 2 * (vv & 3)];
        }
        #pragma unroll
        for (int i = 0; i < 2; ++i)
            #pragma unroll
            for (int j = 0; j < 2; ++j)
                acc[i][j] = __builtin_amdgcn_wmma_f32_16x16x32_f16(
                                false, af[i].h, false, bf[j].h, (short)0, acc[i][j], false, false);
        __syncthreads();
    }

    // epilogue (C layout: VGPR r -> M = half*8 + r, N = r16) — no guards
    #pragma unroll
    for (int i = 0; i < 2; ++i) {
        const int growBase = rowBase + wm * 32 + i * 16 + half * 8;
        #pragma unroll
        for (int j = 0; j < 2; ++j) {
            const int gcol = colBase + wn * 32 + j * 16 + r16;
            const float bv = bias ? bias[gcol] : 0.0f;
            float sc = 1.0f, sh = 0.0f;
            if (bng) {
                float inv = rsqrtf(bnv[gcol] + 1e-5f);
                sc = bng[gcol] * inv;
                sh = bnb[gcol] - bnm[gcol] * sc;
            }
            #pragma unroll
            for (int r = 0; r < 8; ++r) {
                const int grow = growBase + r;
                float fv = acc[i][j][r] + bv;
                if (bng) fv = fv * sc + sh;
                if (act == 1)      fv = fmaxf(fv, 0.0f);
                else if (act == 2) fv = 0.5f * fv * (1.0f + erff(fv * 0.70710678118654752f));
                if (residual) fv += residual[(size_t)grow * N + gcol];
                C[(size_t)grow * N + gcol] = fv;
            }
        }
    }
}

// =====================================================================
// GENERIC WMMA GEMM (edge shapes: K=3, N=40). Tile 64x64, 2 WMMA/wave.
// =====================================================================
__global__ __launch_bounds__(256)
void gemm_wmma_k(const float* __restrict__ A, const _Float16* __restrict__ WT,
                 const float* __restrict__ bias,
                 const float* __restrict__ bng, const float* __restrict__ bnb,
                 const float* __restrict__ bnm, const float* __restrict__ bnv,
                 const float* __restrict__ residual,
                 float* __restrict__ C,
                 int M, int N, int Kd, int Kpad, int act)
{
    __shared__ _Float16 As[64 * 40];
    __shared__ _Float16 Bs[64 * 40];

    const int tid  = threadIdx.x;
    const int lane = tid & 31;
    const int wid  = tid >> 5;
    const int wm   = wid & 3;
    const int wn   = wid >> 2;
    const int half = lane >> 4;
    const int r16  = lane & 15;
    const int rowBase = blockIdx.y * 64;
    const int colBase = blockIdx.x * 64;
    const bool fullN = (colBase + 64 <= N);

    v8f acc0 = {}, acc1 = {};

    for (int kk = 0; kk < Kpad; kk += 32) {
        __syncthreads();
        #pragma unroll
        for (int i = 0; i < 8; ++i) {
            int li = tid + i * 256;
            int r = li >> 5, c = li & 31;
            int gr = rowBase + r, gc = kk + c;
            As[r * 40 + c] = (gr < M && gc < Kd)
                           ? (_Float16)A[(size_t)gr * Kd + gc] : (_Float16)0.0f;
        }
        if (fullN) {
            const int n = tid >> 2, p = (tid & 3) * 8;
            *(uint4*)&Bs[n * 40 + p] =
                *(const uint4*)&WT[(size_t)(colBase + n) * Kpad + kk + p];
        } else {
            #pragma unroll
            for (int i = 0; i < 8; ++i) {
                int li = tid + i * 256;
                int n = li >> 5, c = li & 31;
                int gn = colBase + n;
                Bs[n * 40 + c] = (gn < N) ? WT[(size_t)gn * Kpad + kk + c]
                                          : (_Float16)0.0f;
            }
        }
        __syncthreads();

        union { v16h h; unsigned int u[8]; } af, bf0, bf1;
        const int arow = wm * 16 + r16;
        #pragma unroll
        for (int vv = 0; vv < 4; ++vv)
            af.u[vv] = *(const unsigned int*)&As[arow * 40 + half * 8 + 2 * vv];
        #pragma unroll
        for (int vv = 4; vv < 8; ++vv)
            af.u[vv] = *(const unsigned int*)&As[arow * 40 + 16 + half * 8 + 2 * (vv - 4)];
        const int bc0 = wn * 32 + r16;
        #pragma unroll
        for (int vv = 0; vv < 8; ++vv) {
            bf0.u[vv] = *(const unsigned int*)&Bs[bc0 * 40 + (vv >> 2) * 16 + half * 8 + 2 * (vv & 3)];
            bf1.u[vv] = *(const unsigned int*)&Bs[(bc0 + 16) * 40 + (vv >> 2) * 16 + half * 8 + 2 * (vv & 3)];
        }
        acc0 = __builtin_amdgcn_wmma_f32_16x16x32_f16(
                   false, af.h, false, bf0.h, (short)0, acc0, false, false);
        acc1 = __builtin_amdgcn_wmma_f32_16x16x32_f16(
                   false, af.h, false, bf1.h, (short)0, acc1, false, false);
    }

    #pragma unroll
    for (int sub = 0; sub < 2; ++sub) {
        v8f acc = sub ? acc1 : acc0;
        int gcol = colBase + wn * 32 + sub * 16 + r16;
        if (gcol < N) {
            const float bv = bias ? bias[gcol] : 0.0f;
            float sc = 1.0f, sh = 0.0f;
            if (bng) {
                float inv = rsqrtf(bnv[gcol] + 1e-5f);
                sc = bng[gcol] * inv;
                sh = bnb[gcol] - bnm[gcol] * sc;
            }
            #pragma unroll
            for (int r = 0; r < 8; ++r) {
                int grow = rowBase + wm * 16 + half * 8 + r;
                if (grow < M) {
                    float fv = acc[r] + bv;
                    if (bng) fv = fv * sc + sh;
                    if (act == 1)      fv = fmaxf(fv, 0.0f);
                    else if (act == 2) fv = 0.5f * fv * (1.0f + erff(fv * 0.70710678118654752f));
                    if (residual) fv += residual[(size_t)grow * N + gcol];
                    C[(size_t)grow * N + gcol] = fv;
                }
            }
        }
    }
}

// =====================================================================
// Farthest point sampling: one block per batch, 256 thr x 8 pts.
// =====================================================================
__global__ __launch_bounds__(256)
void fps_k(const float* __restrict__ xyz, float* __restrict__ centers)
{
    const int b = blockIdx.x;
    const float* P = xyz + (size_t)b * NN * 3;
    float px[8], py[8], pz[8], dist[8];
    #pragma unroll
    for (int i = 0; i < 8; ++i) {
        int p = threadIdx.x + i * 256;
        px[i] = P[p * 3 + 0]; py[i] = P[p * 3 + 1]; pz[i] = P[p * 3 + 2];
        dist[i] = 3.4e38f;
    }
    __shared__ float sval[256];
    __shared__ int   sidx[256];
    __shared__ float cc[3];
    int far = 0;
    for (int s = 0; s < GG; ++s) {
        if (threadIdx.x == 0) {
            float cx = P[far * 3], cy = P[far * 3 + 1], cz = P[far * 3 + 2];
            cc[0] = cx; cc[1] = cy; cc[2] = cz;
            float* Co = centers + ((size_t)b * GG + s) * 3;
            Co[0] = cx; Co[1] = cy; Co[2] = cz;
        }
        __syncthreads();
        float cx = cc[0], cy = cc[1], cz = cc[2];
        float best = -1.0f; int bi = 0;
        #pragma unroll
        for (int i = 0; i < 8; ++i) {
            float dx = px[i] - cx, dy = py[i] - cy, dz = pz[i] - cz;
            float d = dx * dx + dy * dy + dz * dz;
            dist[i] = fminf(dist[i], d);
            int p = threadIdx.x + i * 256;
            if (dist[i] > best) { best = dist[i]; bi = p; }
        }
        sval[threadIdx.x] = best; sidx[threadIdx.x] = bi;
        __syncthreads();
        for (int off = 128; off > 0; off >>= 1) {
            if (threadIdx.x < off) {
                float ov = sval[threadIdx.x + off]; int oi = sidx[threadIdx.x + off];
                if (ov > sval[threadIdx.x] ||
                    (ov == sval[threadIdx.x] && oi < sidx[threadIdx.x])) {
                    sval[threadIdx.x] = ov; sidx[threadIdx.x] = oi;
                }
            }
            __syncthreads();
        }
        far = sidx[0];
        __syncthreads();
    }
}

// =====================================================================
// kNN + relative grouping: one block per (b,g). 32 argmin passes.
// =====================================================================
__global__ __launch_bounds__(256)
void knn_k(const float* __restrict__ xyz, const float* __restrict__ centers,
           float* __restrict__ nbrel)
{
    const int blk = blockIdx.x;
    const int b = blk >> 6, g = blk & 63;
    const float* P = xyz + (size_t)b * NN * 3;
    const float* Cc = centers + ((size_t)b * GG + g) * 3;
    const float cx = Cc[0], cy = Cc[1], cz = Cc[2];
    __shared__ float d2[NN];
    __shared__ float sval[256];
    __shared__ int   sidx[256];
    #pragma unroll
    for (int i = 0; i < 8; ++i) {
        int p = threadIdx.x + i * 256;
        float dx = P[p * 3] - cx, dy = P[p * 3 + 1] - cy, dz = P[p * 3 + 2] - cz;
        d2[p] = dx * dx + dy * dy + dz * dz;
    }
    __syncthreads();
    for (int k = 0; k < KK; ++k) {
        float best = 3.5e38f; int bi = 0;
        #pragma unroll
        for (int i = 0; i < 8; ++i) {
            int p = threadIdx.x + i * 256;
            float fv = d2[p];
            if (fv < best) { best = fv; bi = p; }
        }
        sval[threadIdx.x] = best; sidx[threadIdx.x] = bi;
        __syncthreads();
        for (int off = 128; off > 0; off >>= 1) {
            if (threadIdx.x < off) {
                float ov = sval[threadIdx.x + off]; int oi = sidx[threadIdx.x + off];
                if (ov < sval[threadIdx.x] ||
                    (ov == sval[threadIdx.x] && oi < sidx[threadIdx.x])) {
                    sval[threadIdx.x] = ov; sidx[threadIdx.x] = oi;
                }
            }
            __syncthreads();
        }
        if (threadIdx.x == 0) {
            int sel = sidx[0];
            float* o = nbrel + (((size_t)(b * GG + g)) * KK + k) * 3;
            o[0] = P[sel * 3 + 0] - cx;
            o[1] = P[sel * 3 + 1] - cy;
            o[2] = P[sel * 3 + 2] - cz;
            d2[sel] = 3.6e38f;
        }
        __syncthreads();
    }
}

// max over K=32 rows of f2(…,256), write concat [fg | f2] -> cat(…,512)
__global__ __launch_bounds__(256)
void maxcat_k(const float* __restrict__ f2, float* __restrict__ cat)
{
    const int g = blockIdx.x, c = threadIdx.x;
    float m = -3.4e38f;
    for (int k = 0; k < KK; ++k) m = fmaxf(m, f2[((size_t)g * KK + k) * 256 + c]);
    for (int k = 0; k < KK; ++k) {
        size_t r = (size_t)g * KK + k;
        cat[r * 512 + c]       = m;
        cat[r * 512 + 256 + c] = f2[r * 256 + c];
    }
}

// max over K=32 rows of f4(…,384) -> tok(…,384)
__global__ __launch_bounds__(128)
void maxtok_k(const float* __restrict__ f4, float* __restrict__ tok)
{
    const int g = blockIdx.x;
    for (int c = threadIdx.x; c < DD; c += 128) {
        float m = -3.4e38f;
        for (int k = 0; k < KK; ++k) m = fmaxf(m, f4[((size_t)g * KK + k) * DD + c]);
        tok[(size_t)g * DD + c] = m;
    }
}

// prepend cls token / cls pos
__global__ void assemble_k(const float* __restrict__ tok, const float* __restrict__ posc,
                           const float* __restrict__ clsT, const float* __restrict__ clsP,
                           float* __restrict__ x, float* __restrict__ pos)
{
    int i = blockIdx.x * 256 + threadIdx.x;
    if (i >= BB * NTOK * DD) return;
    int r = i / DD, c = i % DD;
    int b = r / NTOK, t = r % NTOK;
    x[i]   = (t == 0) ? clsT[c] : tok[((size_t)b * GG + t - 1) * DD + c];
    pos[i] = (t == 0) ? clsP[c] : posc[((size_t)b * GG + t - 1) * DD + c];
}

__global__ void addpos_k(float* __restrict__ x, const float* __restrict__ pos, int n)
{
    int i = blockIdx.x * 256 + threadIdx.x;
    if (i < n) x[i] += pos[i];
}

// LayerNorm over last dim (384), one block per row
__global__ __launch_bounds__(128)
void layernorm_k(const float* __restrict__ x, const float* __restrict__ gw,
                 const float* __restrict__ bw, float* __restrict__ o, int rows)
{
    const int r = blockIdx.x;
    if (r >= rows) return;
    const int t = threadIdx.x;
    __shared__ float red[128];
    const float* xr = x + (size_t)r * DD;
    float v0 = xr[t], v1 = xr[t + 128], v2 = xr[t + 256];
    red[t] = v0 + v1 + v2;
    __syncthreads();
    for (int off = 64; off > 0; off >>= 1) { if (t < off) red[t] += red[t + off]; __syncthreads(); }
    float mean = red[0] * (1.0f / DD);
    __syncthreads();
    float d0 = v0 - mean, d1 = v1 - mean, d2 = v2 - mean;
    red[t] = d0 * d0 + d1 * d1 + d2 * d2;
    __syncthreads();
    for (int off = 64; off > 0; off >>= 1) { if (t < off) red[t] += red[t + off]; __syncthreads(); }
    float rstd = rsqrtf(red[0] * (1.0f / DD) + 1e-5f);
    float* orow = o + (size_t)r * DD;
    orow[t]       = d0 * rstd * gw[t]       + bw[t];
    orow[t + 128] = d1 * rstd * gw[t + 128] + bw[t + 128];
    orow[t + 256] = d2 * rstd * gw[t + 256] + bw[t + 256];
}

// =====================================================================
// Attention: one block per (b, head), wave32 per query row.
// =====================================================================
__global__ __launch_bounds__(256)
void attention_k(const float* __restrict__ qkv, float* __restrict__ o)
{
    const int bh = blockIdx.x;
    const int b = bh / HH, hh = bh % HH;
    __shared__ float Qs[NTOK * HD];
    __shared__ float Ks[NTOK * HD];
    __shared__ float Vs[NTOK * HD];
    __shared__ float probs[8][72];
    const size_t base = (size_t)b * NTOK * (3 * DD);
    for (int i = threadIdx.x; i < NTOK * HD; i += 256) {
        int t = i >> 6, d = i & 63;
        const float* row = qkv + base + (size_t)t * (3 * DD);
        Qs[i] = row[(0 * HH + hh) * HD + d];
        Ks[i] = row[(1 * HH + hh) * HD + d];
        Vs[i] = row[(2 * HH + hh) * HD + d];
    }
    __syncthreads();
    const int wid = threadIdx.x >> 5, lane = threadIdx.x & 31;
    for (int q = wid; q < NTOK; q += 8) {
        float s0 = 0.0f, s1 = 0.0f, s2 = -1e30f;
        for (int d = 0; d < HD; ++d) {
            float qd = Qs[q * HD + d];
            s0 += qd * Ks[lane * HD + d];
            s1 += qd * Ks[(lane + 32) * HD + d];
        }
        s0 *= ATT_SCALE; s1 *= ATT_SCALE;
        if (lane == 0) {
            float acc = 0.0f;
            for (int d = 0; d < HD; ++d) acc += Qs[q * HD + d] * Ks[64 * HD + d];
            s2 = acc * ATT_SCALE;
        }
        float m = fmaxf(fmaxf(s0, s1), s2);
        for (int off = 16; off > 0; off >>= 1) m = fmaxf(m, __shfl_xor(m, off, 32));
        float e0 = expf(s0 - m), e1 = expf(s1 - m), e2 = expf(s2 - m);
        float sum = e0 + e1 + e2;
        for (int off = 16; off > 0; off >>= 1) sum += __shfl_xor(sum, off, 32);
        float inv = 1.0f / sum;
        probs[wid][lane]      = e0 * inv;
        probs[wid][lane + 32] = e1 * inv;
        if (lane == 0) probs[wid][64] = e2 * inv;
        float o0 = 0.0f, o1 = 0.0f;
        for (int j = 0; j < NTOK; ++j) {
            float p = probs[wid][j];
            o0 += p * Vs[j * HD + lane];
            o1 += p * Vs[j * HD + lane + 32];
        }
        float* orow = o + ((size_t)(b * NTOK + q)) * DD + hh * HD;
        orow[lane]      = o0;
        orow[lane + 32] = o1;
    }
}

// final feature: [cls ; max over tokens 1..64]
__global__ __launch_bounds__(384)
void feat_k(const float* __restrict__ y, float* __restrict__ f)
{
    const int b = blockIdx.x, c = threadIdx.x;
    const float* Y = y + (size_t)b * NTOK * DD;
    f[(size_t)b * (2 * DD) + c] = Y[c];
    float m = -3.4e38f;
    for (int t = 1; t < NTOK; ++t) m = fmaxf(m, Y[(size_t)t * DD + c]);
    f[(size_t)b * (2 * DD) + DD + c] = m;
}

// =====================================================================
// Host side
// =====================================================================
static inline int kpad32(int k) { return (k + 31) & ~31; }

static inline void gemm(hipStream_t s, const float* A, const _Float16* WT, const float* bias,
                        const float* g, const float* bb, const float* m, const float* vv,
                        const float* res, float* C, int M, int N, int K, int act)
{
    if ((M % 64 == 0) && (N % 128 == 0) && (K % 32 == 0)) {
        dim3 grid(N / 128, M / 64);
        gemm_wmma_fast_k<<<grid, dim3(256), 0, s>>>(A, WT, bias, g, bb, m, vv, res, C,
                                                    M, N, K, act);
    } else {
        dim3 grid((N + 63) / 64, (M + 63) / 64);
        gemm_wmma_k<<<grid, dim3(256), 0, s>>>(A, WT, bias, g, bb, m, vv, res, C,
                                               M, N, K, kpad32(K), act);
    }
}

extern "C" void kernel_launch(void* const* d_in, const int* in_sizes, int n_in,
                              void* d_out, int out_size, void* d_ws, size_t ws_size,
                              hipStream_t stream)
{
    (void)in_sizes; (void)n_in; (void)out_size; (void)ws_size;
    int ix = 0;
    const float* xyz = (const float*)d_in[ix++];   // (B,N,3)
    // encoder
    const float* e_w1 = (const float*)d_in[ix++]; const float* e_b1 = (const float*)d_in[ix++];
    const float* e_bn1g = (const float*)d_in[ix++]; const float* e_bn1b = (const float*)d_in[ix++];
    const float* e_bn1m = (const float*)d_in[ix++]; const float* e_bn1v = (const float*)d_in[ix++];
    const float* e_w2 = (const float*)d_in[ix++]; const float* e_b2 = (const float*)d_in[ix++];
    const float* e_w3 = (const float*)d_in[ix++]; const float* e_b3 = (const float*)d_in[ix++];
    const float* e_bn2g = (const float*)d_in[ix++]; const float* e_bn2b = (const float*)d_in[ix++];
    const float* e_bn2m = (const float*)d_in[ix++]; const float* e_bn2v = (const float*)d_in[ix++];
    const float* e_w4 = (const float*)d_in[ix++]; const float* e_b4 = (const float*)d_in[ix++];
    // pos mlp
    const float* p_w1 = (const float*)d_in[ix++]; const float* p_b1 = (const float*)d_in[ix++];
    const float* p_w2 = (const float*)d_in[ix++]; const float* p_b2 = (const float*)d_in[ix++];
    const float* cls_tok = (const float*)d_in[ix++];
    const float* cls_pos = (const float*)d_in[ix++];
    // blocks (stacked over DEPTH)
    const float* ln1_g = (const float*)d_in[ix++]; const float* ln1_b = (const float*)d_in[ix++];
    const float* qkv_w = (const float*)d_in[ix++];
    const float* proj_w = (const float*)d_in[ix++]; const float* proj_b = (const float*)d_in[ix++];
    const float* ln2_g = (const float*)d_in[ix++]; const float* ln2_b = (const float*)d_in[ix++];
    const float* fc1_w = (const float*)d_in[ix++]; const float* fc1_b = (const float*)d_in[ix++];
    const float* fc2_w = (const float*)d_in[ix++]; const float* fc2_b = (const float*)d_in[ix++];
    const float* norm_g = (const float*)d_in[ix++]; const float* norm_b = (const float*)d_in[ix++];
    // head
    const float* h_w1 = (const float*)d_in[ix++]; const float* h_b1 = (const float*)d_in[ix++];
    const float* h_bn1g = (const float*)d_in[ix++]; const float* h_bn1b = (const float*)d_in[ix++];
    const float* h_bn1m = (const float*)d_in[ix++]; const float* h_bn1v = (const float*)d_in[ix++];
    const float* h_w2 = (const float*)d_in[ix++]; const float* h_b2 = (const float*)d_in[ix++];
    const float* h_bn2g = (const float*)d_in[ix++]; const float* h_bn2b = (const float*)d_in[ix++];
    const float* h_bn2m = (const float*)d_in[ix++]; const float* h_bn2v = (const float*)d_in[ix++];
    const float* h_w3 = (const float*)d_in[ix++]; const float* h_b3 = (const float*)d_in[ix++];

    // ---------------- workspace carve (256B aligned) ----------------
    char* base = (char*)d_ws;
    size_t off = 0;
    auto carveB = [&](size_t bytes) -> char* {
        size_t o = (off + 255) & ~(size_t)255;
        off = o + bytes;
        return base + o;
    };
    auto carveF = [&](size_t n) { return (float*)carveB(n * sizeof(float)); };
    auto carveH = [&](size_t n) { return (_Float16*)carveB(n * sizeof(_Float16)); };

    const int BG = BB * GG;                 // 4096 groups
    const int NROW = BB * NTOK;             // 4160 token rows
    float* centers = carveF((size_t)BG * 3);
    float* nbrel   = carveF((size_t)BG * KK * 3);
    float* posh    = carveF((size_t)BG * 128);
    float* posc    = carveF((size_t)BG * DD);
    float* tok     = carveF((size_t)BG * DD);
    float* x       = carveF((size_t)NROW * DD);
    float* pos     = carveF((size_t)NROW * DD);
    float* hbuf    = carveF((size_t)NROW * DD);
    float* qkv     = carveF((size_t)NROW * 3 * DD);
    float* attno   = carveF((size_t)NROW * DD);
    float* t1      = carveF((size_t)NROW * 4 * DD);
    float* fvec    = carveF((size_t)BB * 2 * DD);
    float* hh1     = carveF((size_t)BB * 256);
    float* hh2     = carveF((size_t)BB * 256);
    // encoder chunk buffers: 256 groups -> 8192 rows
    const int CHG = 256, CHR = CHG * KK;    // 8192
    float* e1   = carveF((size_t)CHR * 128);
    float* e2   = carveF((size_t)CHR * 256);
    float* ecat = carveF((size_t)CHR * 512);
    float* e3   = carveF((size_t)CHR * 512);
    float* e4   = carveF((size_t)CHR * DD);

    // ---------------- weight pre-transpose to f16 ----------------
    auto xpose = [&](const float* W, int K, int N) -> _Float16* {
        int Kp = kpad32(K);
        _Float16* WT = carveH((size_t)N * Kp);
        int total = N * Kp;
        cvt_wt_k<<<(total + 255) / 256, 256, 0, stream>>>(W, WT, K, N, Kp);
        return WT;
    };
    _Float16* wt_ew1 = xpose(e_w1, 3, 128);
    _Float16* wt_ew2 = xpose(e_w2, 128, 256);
    _Float16* wt_ew3 = xpose(e_w3, 512, 512);
    _Float16* wt_ew4 = xpose(e_w4, 512, DD);
    _Float16* wt_pw1 = xpose(p_w1, 3, 128);
    _Float16* wt_pw2 = xpose(p_w2, 128, DD);
    _Float16* wt_qkv[DEPTH]; _Float16* wt_proj[DEPTH];
    _Float16* wt_fc1[DEPTH]; _Float16* wt_fc2[DEPTH];
    for (int d = 0; d < DEPTH; ++d) {
        wt_qkv[d]  = xpose(qkv_w  + (size_t)d * DD * 3 * DD, DD, 3 * DD);
        wt_proj[d] = xpose(proj_w + (size_t)d * DD * DD,     DD, DD);
        wt_fc1[d]  = xpose(fc1_w  + (size_t)d * DD * 4 * DD, DD, 4 * DD);
        wt_fc2[d]  = xpose(fc2_w  + (size_t)d * 4 * DD * DD, 4 * DD, DD);
    }
    _Float16* wt_hw1 = xpose(h_w1, 2 * DD, 256);
    _Float16* wt_hw2 = xpose(h_w2, 256, 256);
    _Float16* wt_hw3 = xpose(h_w3, 256, NCLS);

    // ---------------- grouping ----------------
    fps_k<<<BB, 256, 0, stream>>>(xyz, centers);
    knn_k<<<BB * GG, 256, 0, stream>>>(xyz, centers, nbrel);

    // ---------------- encoder (chunked over groups) ----------------
    for (int ci = 0; ci < BG / CHG; ++ci) {
        const float* Ain = nbrel + (size_t)ci * CHR * 3;
        gemm(stream, Ain, wt_ew1, e_b1, e_bn1g, e_bn1b, e_bn1m, e_bn1v, nullptr,
             e1, CHR, 128, 3, 1);
        gemm(stream, e1, wt_ew2, e_b2, nullptr, nullptr, nullptr, nullptr, nullptr,
             e2, CHR, 256, 128, 0);
        maxcat_k<<<CHG, 256, 0, stream>>>(e2, ecat);
        gemm(stream, ecat, wt_ew3, e_b3, e_bn2g, e_bn2b, e_bn2m, e_bn2v, nullptr,
             e3, CHR, 512, 512, 1);
        gemm(stream, e3, wt_ew4, e_b4, nullptr, nullptr, nullptr, nullptr, nullptr,
             e4, CHR, DD, 512, 0);
        maxtok_k<<<CHG, 128, 0, stream>>>(e4, tok + (size_t)ci * CHG * DD);
    }

    // ---------------- positional embedding ----------------
    gemm(stream, centers, wt_pw1, p_b1, nullptr, nullptr, nullptr, nullptr, nullptr,
         posh, BG, 128, 3, 2);
    gemm(stream, posh, wt_pw2, p_b2, nullptr, nullptr, nullptr, nullptr, nullptr,
         posc, BG, DD, 128, 0);

    assemble_k<<<(NROW * DD + 255) / 256, 256, 0, stream>>>(tok, posc, cls_tok, cls_pos, x, pos);

    // ---------------- transformer ----------------
    const int nelem = NROW * DD;
    for (int d = 0; d < DEPTH; ++d) {
        addpos_k<<<(nelem + 255) / 256, 256, 0, stream>>>(x, pos, nelem);
        layernorm_k<<<NROW, 128, 0, stream>>>(x, ln1_g + (size_t)d * DD, ln1_b + (size_t)d * DD, hbuf, NROW);
        gemm(stream, hbuf, wt_qkv[d], nullptr,
             nullptr, nullptr, nullptr, nullptr, nullptr, qkv, NROW, 3 * DD, DD, 0);
        attention_k<<<BB * HH, 256, 0, stream>>>(qkv, attno);
        gemm(stream, attno, wt_proj[d], proj_b + (size_t)d * DD,
             nullptr, nullptr, nullptr, nullptr, x, x, NROW, DD, DD, 0);
        layernorm_k<<<NROW, 128, 0, stream>>>(x, ln2_g + (size_t)d * DD, ln2_b + (size_t)d * DD, hbuf, NROW);
        gemm(stream, hbuf, wt_fc1[d], fc1_b + (size_t)d * 4 * DD,
             nullptr, nullptr, nullptr, nullptr, nullptr, t1, NROW, 4 * DD, DD, 2);
        gemm(stream, t1, wt_fc2[d], fc2_b + (size_t)d * DD,
             nullptr, nullptr, nullptr, nullptr, x, x, NROW, DD, 4 * DD, 0);
    }

    // ---------------- head ----------------
    layernorm_k<<<NROW, 128, 0, stream>>>(x, norm_g, norm_b, hbuf, NROW);
    feat_k<<<BB, 384, 0, stream>>>(hbuf, fvec);
    gemm(stream, fvec, wt_hw1, h_b1, h_bn1g, h_bn1b, h_bn1m, h_bn1v, nullptr,
         hh1, BB, 256, 2 * DD, 1);
    gemm(stream, hh1, wt_hw2, h_b2, h_bn2g, h_bn2b, h_bn2m, h_bn2v, nullptr,
         hh2, BB, 256, 256, 1);
    gemm(stream, hh2, wt_hw3, h_b3, nullptr, nullptr, nullptr, nullptr, nullptr,
         (float*)d_out, BB, NCLS, 256, 0);
}
